// SpMiddleFHDPeople_197568495910
// MI455X (gfx1250) — matrix operand order repaired and checked by reading, not compiled, not run on hardware
//
#include <hip/hip_runtime.h>
#include <stdint.h>
#include <stddef.h>

// ---------------------------------------------------------------------------
// SpMiddleFHD middle_conv for MI455X (gfx1250), bf16 WMMA implicit-GEMM.
//
// Roofline: ~84 GFLOP dense; densified bf16 activation grid (137MB) fits in
// the 192MB L2, so the gather-heavy implicit GEMM is L2-resident and the
// kernel is WMMA-issue bound. bf16 (f32 exponent range) in, f32 accumulate.
// BN scale folded into weights at repack time; bias+ReLU+mask in epilogue.
// Per-layer packed weights are DMA'd into LDS once per workgroup by the
// Tensor Data Mover (TENSORcnt-tracked), B fragments then come from LDS.
//
// d_in layout assumption (setup_inputs dict order, tuples flattened):
//   [0] voxel_features (NV*128 f32)   [1] coors (NV*4 i32)   [2] batch_size
//   [3..12] weights, [13..22] gammas, [23..32] betas, [33..42] means,
//   [43..52] vars.
// ---------------------------------------------------------------------------

typedef __attribute__((ext_vector_type(16))) __bf16 v16bf;
typedef __attribute__((ext_vector_type(8)))  float  v8f;
typedef __attribute__((ext_vector_type(4)))  unsigned int v4u;
typedef __attribute__((ext_vector_type(4)))  unsigned int u32x4;
typedef __attribute__((ext_vector_type(8)))  int i32x8;
typedef __attribute__((ext_vector_type(4)))  int i32x4;

#define EPS_BN 1e-3f

__device__ __forceinline__ unsigned short f32_to_bf16(float f) {
  unsigned int u = __float_as_uint(f);
  return (unsigned short)((u + 0x7FFFu + ((u >> 16) & 1u)) >> 16);  // RNE
}
__device__ __forceinline__ float bf16_to_f32(unsigned short h) {
  return __uint_as_float(((unsigned int)h) << 16);
}

// ---------------- zero fill (16B stores) ----------------
__global__ void zero_v4_kernel(v4u* dst, int n) {
  v4u z = {0u, 0u, 0u, 0u};
  for (int i = blockIdx.x * blockDim.x + threadIdx.x; i < n;
       i += gridDim.x * blockDim.x)
    dst[i] = z;
}

// ---------------- scatter voxels into bf16 grid + mask ----------------
__global__ void scatter_kernel(const float* __restrict__ vf,
                               const int* __restrict__ coors,
                               unsigned short* __restrict__ g0,
                               unsigned char* __restrict__ m0, int nv) {
  int t = blockIdx.x * blockDim.x + threadIdx.x;
  if (t >= nv * 128) return;
  int v = t >> 7, c = t & 127;
  int z = coors[v * 4 + 1], y = coors[v * 4 + 2], x = coors[v * 4 + 3];
  int pos = (z * 160 + y) * 160 + x;
  g0[pos * 128 + c] = f32_to_bf16(vf[t]);
  if (c == 0) m0[pos] = 1;
}

// ---------------- weight repack into WMMA B-fragment order ----------------
// Packed half index: (((tap*KB + kb)*NT + nt)*32 + lane)*16 + i
// holds weight(out_ch = nt*16 + lane%16,
//              in_ch  = kb*32 + (i<8 ? 8*(lane/16)+i : 16+8*(lane/16)+i-8), tap) * bn_scale.
// (16-bit dense B 32x16: lane l -> column N=l%16, halves K={0..7}U{16..23} (l<16) or {8..15}U{24..31})
struct RepackParams {
  const float* w; const float* gamma; const float* var;
  unsigned short* dst;
  int Cin, Cout, kd, kh, kw, KB, NT, total;
};
__global__ void repack_kernel(RepackParams p) {
  int t = blockIdx.x * blockDim.x + threadIdx.x;
  if (t >= p.total) return;
  int i = t & 15;
  int lane = (t >> 4) & 31;
  int r = t >> 9;
  int nt = r % p.NT; r /= p.NT;
  int kb = r % p.KB; int tap = r / p.KB;
  int n = (nt << 4) + (lane & 15);
  int k = (i < 8 ? 0 : 8) + (lane < 16 ? 0 : 8) + i;
  int ic = (kb << 5) + k;
  float val = 0.0f;  // zero-pad K when Cin < KB*32 (e.g. Cin=16)
  if (ic < p.Cin) {
    int zw = tap % p.kw;
    int zh = (tap / p.kw) % p.kh;
    int zd = tap / (p.kw * p.kh);
    float s = p.gamma[n] * rsqrtf(p.var[n] + EPS_BN);
    val = p.w[(((n * p.Cin + ic) * p.kd + zd) * p.kh + zh) * p.kw + zw] * s;
  }
  p.dst[t] = f32_to_bf16(val);
}

__global__ void bias_kernel(const float* gamma, const float* beta,
                            const float* mean, const float* var,
                            float* dst, int Co) {
  int n = blockIdx.x * blockDim.x + threadIdx.x;
  if (n >= Co) return;
  float s = gamma[n] * rsqrtf(var[n] + EPS_BN);
  dst[n] = beta[n] - mean[n] * s;
}

// ---------------- sparse-conv output mask dilation ----------------
struct MaskParams {
  const unsigned char* min_;
  unsigned char* mout;
  int Din, Hin, Win, Dout, Hout, Wout, kd, kh, kw, sd, sh, sw, pd, ph, pw;
};
__global__ void mask_pool_kernel(MaskParams p) {
  int t = blockIdx.x * blockDim.x + threadIdx.x;
  int total = p.Dout * p.Hout * p.Wout;
  if (t >= total) return;
  int w = t % p.Wout;
  int h = (t / p.Wout) % p.Hout;
  int d = t / (p.Wout * p.Hout);
  unsigned char any = 0;
  for (int zd = 0; zd < p.kd; ++zd) {
    int id = d * p.sd + zd - p.pd;
    if ((unsigned)id >= (unsigned)p.Din) continue;
    for (int zh = 0; zh < p.kh; ++zh) {
      int ih = h * p.sh + zh - p.ph;
      if ((unsigned)ih >= (unsigned)p.Hin) continue;
      for (int zw = 0; zw < p.kw; ++zw) {
        int iw = w * p.sw + zw - p.pw;
        if ((unsigned)iw >= (unsigned)p.Win) continue;
        any |= p.min_[(id * p.Hin + ih) * p.Win + iw];
      }
    }
  }
  p.mout[t] = any ? 1 : 0;
}

// ---------------- implicit-GEMM conv + BN + ReLU + mask (bf16 WMMA) --------
// One wave = one M-tile (16 consecutive W outputs at fixed od,oh) x all Cout.
// A fragment (16-bit 16x32): lane<16 holds M=lane, K {0..7}U{16..23};
// lane>=16 holds M=lane-16, K {8..15}U{24..31} -> two b128 loads per tap/kb.
// B fragments are served from LDS after a one-shot TDM tensor_load_to_lds.
struct ConvParams {
  const unsigned short* in;   // [Din*Hin*Win, Cin] bf16 channels-last
  unsigned short* out;        // [Dout*Hout*Wout, Cout] bf16
  const unsigned short* wp;   // packed weights (global)
  const float* bias;
  const unsigned char* mask;  // out-resolution active mask
  int Din, Hin, Win, Cin;
  int Dout, Hout, Wout, Cout;
  int kd, kh, kw, sd, sh, sw, pd, ph, pw;
  int TW;                     // ceil(Wout/16)
  int whalves;                // packed weight halves for this layer (mult of 512)
};

template <int KB, int NT>
__global__ __launch_bounds__(128) void conv_wmma_kernel(ConvParams p) {
  extern __shared__ __align__(16) unsigned char smem[];

  // ---- Stage this layer's packed weights into LDS (TDM, TENSORcnt). ----
#if __has_builtin(__builtin_amdgcn_tensor_load_to_lds)
  if ((threadIdx.x >> 5) == 0) {  // one TDM issue per workgroup (wave 0)
    const unsigned long long ga = (unsigned long long)(uintptr_t)p.wp;
    const int rows = p.whalves >> 9;  // rows of 512 halves (1KB each)
    u32x4 g0;
    g0.x = 1u;                                     // count=1, valid user D#
    g0.y = (unsigned)(uintptr_t)smem;              // lds_addr (bytes)
    g0.z = (unsigned)(ga & 0xFFFFFFFFull);         // global_addr[31:0]
    g0.w = (unsigned)((ga >> 32) & 0x01FFFFFFull)  // global_addr[56:32]
           | (2u << 30);                           // type = 2 ("image")
    i32x8 g1;
    g1[0] = 0x00010000;             // wg_mask=0, data_size=1 (2 bytes)
    g1[1] = 512 << 16;              // tensor_dim0 = 512 (bits 79:48, low16)
    g1[2] = (rows & 0xFFFF) << 16;  // tensor_dim1 = rows (bits 111:80, low16)
    g1[3] = 512 << 16;              // tile_dim0 = 512 (bits 127:112)
    g1[4] = rows & 0xFFFF;          // tile_dim1 = rows (bits 143:128)
    g1[5] = 512;                    // tensor_dim0_stride = 512 (bits 207:160)
    g1[6] = 0;
    g1[7] = 0;                      // tensor_dim1_stride = 0 (2D tile)
    i32x4 gz = {0, 0, 0, 0};
#if __clang_major__ >= 23
    i32x8 gz8 = {0, 0, 0, 0, 0, 0, 0, 0};
    __builtin_amdgcn_tensor_load_to_lds(g0, g1, gz, gz, gz8, 0);
#else
    __builtin_amdgcn_tensor_load_to_lds(g0, g1, gz, gz, 0);
#endif
#if __has_builtin(__builtin_amdgcn_s_wait_tensorcnt)
    __builtin_amdgcn_s_wait_tensorcnt(0);
#endif
  }
#else
  // Fallback: cooperative global->LDS copy.
  for (int i = threadIdx.x; i < (p.whalves >> 3); i += blockDim.x)
    ((v4u*)smem)[i] = ((const v4u*)p.wp)[i];
#endif
  __syncthreads();

  const int wave = blockIdx.x * (blockDim.x >> 5) + (threadIdx.x >> 5);
  const int lane = threadIdx.x & 31;
  const int total = p.Dout * p.Hout * p.TW;
  if (wave >= total) return;  // wave-uniform exit (after the barrier)

  const int tw = wave % p.TW;
  const int oh = (wave / p.TW) % p.Hout;
  const int od = wave / (p.TW * p.Hout);
  const int owbase = tw << 4;
  const int mrow = lane & 15;
  const int ow = owbase + mrow;
  const bool hi = lane >= 16;
  const int iw0 = ow * p.sw - p.pw;
  const int ih0 = oh * p.sh - p.ph;
  const int id0 = od * p.sd - p.pd;

  v8f zacc = {0.f, 0.f, 0.f, 0.f, 0.f, 0.f, 0.f, 0.f};
  v8f acc[NT];
#pragma unroll
  for (int nt = 0; nt < NT; ++nt) acc[nt] = zacc;

  const v4u* inq = (const v4u*)p.in;
  const v4u* wq = (const v4u*)smem;  // LDS-resident packed weights
  const int cq = p.Cin >> 3;         // v4u (8 halves) per input position

  int tap = 0;
  for (int zd = 0; zd < p.kd; ++zd) {
    const int id = id0 + zd;
    const bool dok = (unsigned)id < (unsigned)p.Din;
    for (int zh = 0; zh < p.kh; ++zh) {
      const int ih = ih0 + zh;
      const bool hok = dok && ((unsigned)ih < (unsigned)p.Hin);
      const int rowb = (id * p.Hin + ih) * p.Win;
      for (int zw = 0; zw < p.kw; ++zw, ++tap) {
        const int iw = iw0 + zw;
        const bool ok =
            hok && ((unsigned)iw < (unsigned)p.Win) && (ow < p.Wout);
        const int aq0 = (rowb + iw) * cq + (hi ? 1 : 0);
        // Speculative prefetch of the next tap's input row (L2-resident).
        __builtin_prefetch(inq + aq0 + cq, 0, 1);
#pragma unroll
        for (int kb = 0; kb < KB; ++kb) {
          union { v4u q[2]; v16bf v; } a;
          v4u z4 = {0u, 0u, 0u, 0u};
          a.q[0] = z4;
          a.q[1] = z4;
          if (ok) {  // zero-padding at volume edges
            const int qi = aq0 + (kb << 2);
            a.q[0] = inq[qi];      // K half-chunk {0..7} or {8..15}
            a.q[1] = inq[qi + 2];  // K half-chunk {16..23} or {24..31}
          }
          const int wb = ((tap * KB + kb) * NT) * 64 + (lane << 1);
#pragma unroll
          for (int nt = 0; nt < NT; ++nt) {
            union { v4u q[2]; v16bf v; } b;
            b.q[0] = wq[wb + nt * 64];      // ds_load_b128
            b.q[1] = wq[wb + nt * 64 + 1];  // ds_load_b128
            acc[nt] = __builtin_amdgcn_wmma_f32_16x16x32_bf16(
                false, a.v, false, b.v, (short)0, acc[nt], false, false);
          }
        }
      }
    }
  }

  // Epilogue: C/D layout is VGPR r -> M = r + 8*(lane>=16), N = lane%16.
  const int opb = (od * p.Hout + oh) * p.Wout;
  const int n0 = lane & 15;
  const int mbase = hi ? 8 : 0;
  float mk[8];
#pragma unroll
  for (int r = 0; r < 8; ++r) {
    const int owv = owbase + mbase + r;
    mk[r] = (owv < p.Wout) ? (float)p.mask[opb + owv] : 0.0f;
  }
#pragma unroll
  for (int nt = 0; nt < NT; ++nt) {
    const int n = (nt << 4) + n0;
    const float bs = p.bias[n];
#pragma unroll
    for (int r = 0; r < 8; ++r) {
      const int owv = owbase + mbase + r;
      if (owv < p.Wout) {
        const int pos = opb + owv;
        float v = acc[nt][r] + bs;
        v = fmaxf(v, 0.0f);
        v *= mk[r];
        p.out[pos * p.Cout + n] = f32_to_bf16(v);
      }
    }
  }
}

// ---------------- final reshape [1,64,2,40,40] -> [1,128,40,40] f32 --------
__global__ void gather_out_kernel(const unsigned short* __restrict__ g,
                                  float* __restrict__ out) {
  int t = blockIdx.x * blockDim.x + threadIdx.x;
  if (t >= 128 * 40 * 40) return;
  int w = t % 40;
  int h = (t / 40) % 40;
  int cd = t / 1600;          // out channel index = c*D + d, D=2
  int c = cd >> 1, d = cd & 1;
  out[t] = bf16_to_f32(g[((d * 40 + h) * 40 + w) * 64 + c]);
}

// ---------------------------------------------------------------------------
// Host-side orchestration
// ---------------------------------------------------------------------------
struct LDef {
  int Cin, Cout, Din, Hin, Win, Dout, Hout, Wout;
  int kd, kh, kw, sd, sh, sw, pd, ph, pw, sp;
};
static const LDef g_layers[10] = {
    {128, 16, 21, 160, 160, 21, 160, 160, 3, 3, 3, 1, 1, 1, 1, 1, 1, 0},
    { 16, 16, 21, 160, 160, 21, 160, 160, 3, 3, 3, 1, 1, 1, 1, 1, 1, 0},
    { 16, 32, 21, 160, 160, 11,  80,  80, 3, 3, 3, 2, 2, 2, 1, 1, 1, 1},
    { 32, 32, 11,  80,  80, 11,  80,  80, 3, 3, 3, 1, 1, 1, 1, 1, 1, 0},
    { 32, 32, 11,  80,  80, 11,  80,  80, 3, 3, 3, 1, 1, 1, 1, 1, 1, 0},
    { 32, 64, 11,  80,  80,  5,  40,  40, 3, 3, 3, 2, 2, 2, 0, 1, 1, 1},
    { 64, 64,  5,  40,  40,  5,  40,  40, 3, 3, 3, 1, 1, 1, 1, 1, 1, 0},
    { 64, 64,  5,  40,  40,  5,  40,  40, 3, 3, 3, 1, 1, 1, 1, 1, 1, 0},
    { 64, 64,  5,  40,  40,  5,  40,  40, 3, 3, 3, 1, 1, 1, 1, 1, 1, 0},
    { 64, 64,  5,  40,  40,  2,  40,  40, 3, 1, 1, 2, 1, 1, 0, 0, 0, 1},
};

static inline size_t alup(size_t x) { return (x + 255) & ~(size_t)255; }

static void launch_conv(const ConvParams& cp, int KB, int NT,
                        hipStream_t stream) {
  int waves = cp.Dout * cp.Hout * cp.TW;
  int blocks = (waves + 3) / 4;            // 128 threads = 4 wave32 per block
  size_t shb = (size_t)cp.whalves * 2;     // dynamic LDS for packed weights
  if (KB == 4 && NT == 1)
    conv_wmma_kernel<4, 1><<<blocks, 128, shb, stream>>>(cp);
  else if (KB == 1 && NT == 1)
    conv_wmma_kernel<1, 1><<<blocks, 128, shb, stream>>>(cp);
  else if (KB == 1 && NT == 2)
    conv_wmma_kernel<1, 2><<<blocks, 128, shb, stream>>>(cp);
  else if (KB == 1 && NT == 4)
    conv_wmma_kernel<1, 4><<<blocks, 128, shb, stream>>>(cp);
  else if (KB == 2 && NT == 4)
    conv_wmma_kernel<2, 4><<<blocks, 128, shb, stream>>>(cp);
}

extern "C" void kernel_launch(void* const* d_in, const int* in_sizes, int n_in,
                              void* d_out, int out_size, void* d_ws,
                              size_t ws_size, hipStream_t stream) {
  (void)n_in; (void)out_size; (void)ws_size;
  const float* vf = (const float*)d_in[0];
  const int* coors = (const int*)d_in[1];
  const int nv = in_sizes[0] / 128;
  const float *W[10], *GM[10], *BT[10], *MN[10], *VR[10];
  for (int i = 0; i < 10; ++i) {
    W[i] = (const float*)d_in[3 + i];
    GM[i] = (const float*)d_in[13 + i];
    BT[i] = (const float*)d_in[23 + i];
    MN[i] = (const float*)d_in[33 + i];
    VR[i] = (const float*)d_in[43 + i];
  }

  // ---- workspace carve (~175MB) ----
  char* ws = (char*)d_ws;
  size_t off = 0;
  unsigned short* G0 = (unsigned short*)(ws + off);           // 128ch full res
  off = alup(off + (size_t)537600 * 128 * 2 + 256);
  unsigned short* GA = (unsigned short*)(ws + off);           // ping
  off = alup(off + (size_t)537600 * 16 * 2 + 256);
  unsigned short* GB = (unsigned short*)(ws + off);           // pong
  off = alup(off + (size_t)537600 * 16 * 2 + 256);
  unsigned char* MB[4];
  const size_t msz[4] = {537600, 70400, 8000, 3200};
  for (int i = 0; i < 4; ++i) {
    MB[i] = (unsigned char*)(ws + off);
    off = alup(off + msz[i] + 256);
  }
  unsigned short* WP = (unsigned short*)(ws + off);
  int wofh[10], wsz[10], boff[10];
  {
    int a = 0, b = 0;
    for (int i = 0; i < 10; ++i) {
      const LDef& L = g_layers[i];
      int taps = L.kd * L.kh * L.kw;
      int KB = (L.Cin + 31) / 32, NT = L.Cout / 16;
      wofh[i] = a;
      wsz[i] = taps * KB * NT * 512;
      a += wsz[i];
      boff[i] = b;
      b += L.Cout;
    }
    off = alup(off + (size_t)a * 2 + 4096);
  }
  float* BIAS = (float*)(ws + off);
  off = alup(off + 448 * 4 + 256);

  // 1) zero densified input grid + full-res active mask
  zero_v4_kernel<<<2048, 256, 0, stream>>>((v4u*)G0, 537600 * 128 * 2 / 16);
  zero_v4_kernel<<<64, 256, 0, stream>>>((v4u*)MB[0], 537600 / 16);

  // 2) repack weights (fold BN scale) + biases
  for (int i = 0; i < 10; ++i) {
    const LDef& L = g_layers[i];
    RepackParams rp;
    rp.w = W[i]; rp.gamma = GM[i]; rp.var = VR[i];
    rp.dst = WP + wofh[i];
    rp.Cin = L.Cin; rp.Cout = L.Cout;
    rp.kd = L.kd; rp.kh = L.kh; rp.kw = L.kw;
    rp.KB = (L.Cin + 31) / 32; rp.NT = L.Cout / 16;
    rp.total = wsz[i];
    repack_kernel<<<(wsz[i] + 255) / 256, 256, 0, stream>>>(rp);
    bias_kernel<<<1, 64, 0, stream>>>(GM[i], BT[i], MN[i], VR[i],
                                      BIAS + boff[i], L.Cout);
  }

  // 3) scatter voxel features
  scatter_kernel<<<(nv * 128 + 255) / 256, 256, 0, stream>>>(vf, coors, G0,
                                                             MB[0], nv);

  // 4) the 10 conv layers (mask dilation before each strided sparse conv)
  const unsigned short* cin = G0;
  unsigned short* last = GB;
  int mi = 0;
  for (int i = 0; i < 10; ++i) {
    const LDef& L = g_layers[i];
    unsigned short* cout_ = (i & 1) ? GB : GA;
    if (L.sp) {
      MaskParams mp;
      mp.min_ = MB[mi]; mp.mout = MB[mi + 1];
      mp.Din = L.Din; mp.Hin = L.Hin; mp.Win = L.Win;
      mp.Dout = L.Dout; mp.Hout = L.Hout; mp.Wout = L.Wout;
      mp.kd = L.kd; mp.kh = L.kh; mp.kw = L.kw;
      mp.sd = L.sd; mp.sh = L.sh; mp.sw = L.sw;
      mp.pd = L.pd; mp.ph = L.ph; mp.pw = L.pw;
      int tot = L.Dout * L.Hout * L.Wout;
      mask_pool_kernel<<<(tot + 255) / 256, 256, 0, stream>>>(mp);
      ++mi;
    }
    ConvParams cp;
    cp.in = cin; cp.out = cout_;
    cp.wp = WP + wofh[i]; cp.bias = BIAS + boff[i]; cp.mask = MB[mi];
    cp.Din = L.Din; cp.Hin = L.Hin; cp.Win = L.Win; cp.Cin = L.Cin;
    cp.Dout = L.Dout; cp.Hout = L.Hout; cp.Wout = L.Wout; cp.Cout = L.Cout;
    cp.kd = L.kd; cp.kh = L.kh; cp.kw = L.kw;
    cp.sd = L.sd; cp.sh = L.sh; cp.sw = L.sw;
    cp.pd = L.pd; cp.ph = L.ph; cp.pw = L.pw;
    cp.TW = (L.Wout + 15) / 16;
    cp.whalves = wsz[i];
    launch_conv(cp, (L.Cin + 31) / 32, L.Cout / 16, stream);
    cin = cout_;
    last = cout_;
  }

  // 5) reshape/convert [2,40,40,64] bf16 -> [128,40,40] f32
  gather_out_kernel<<<(204800 + 255) / 256, 256, 0, stream>>>(last,
                                                              (float*)d_out);
}
